// TemporalAdapter_27367531610970
// MI455X (gfx1250) — compile-verified, matrix operand order checked by reading, not run on hardware
//
#include <hip/hip_runtime.h>
#include <hip/hip_bf16.h>

typedef __attribute__((ext_vector_type(16))) _Float16 v16h;
typedef __attribute__((ext_vector_type(8)))  _Float16 v8h;
typedef __attribute__((ext_vector_type(8)))  float    v8f;

#define LSEQ 1024
#define WDIM 1024
#define FDIM 4096

// ---------------------------------------------------------------------------
// CDNA5 async global->LDS (ASYNCcnt-tracked), 16B per lane.
// lds_off = wave-relative LDS byte address (low 32 bits of generic pointer).
// ---------------------------------------------------------------------------
__device__ __forceinline__ void async_b128(unsigned int lds_off, const void* g) {
    asm volatile("global_load_async_to_lds_b128 %0, %1, off"
                 :: "v"(lds_off), "v"((unsigned long long)(uintptr_t)g)
                 : "memory");
}
__device__ __forceinline__ void wait_async0() {
    asm volatile("s_wait_asynccnt 0x0" ::: "memory");
}

// ---------------------------------------------------------------------------
// attn[i][j] = exp(dist[i][j]) / sum_j exp(dist[i][j]),  dist = exp(-10|i-j|)
// ---------------------------------------------------------------------------
__global__ __launch_bounds__(256) void attn_kernel(_Float16* __restrict__ attn) {
    const int i = blockIdx.x;
    const int t = threadIdx.x;
    __shared__ float sred[256];
    float e[4];
    float s = 0.0f;
#pragma unroll
    for (int c = 0; c < 4; ++c) {
        int j = t + c * 256;
        float d = __expf(-10.0f * fabsf((float)(i - j)));
        e[c] = __expf(d);
        s += e[c];
    }
    sred[t] = s;
    __syncthreads();
    for (int stp = 128; stp > 0; stp >>= 1) {
        if (t < stp) sred[t] += sred[t + stp];
        __syncthreads();
    }
    const float inv = 1.0f / sred[0];
#pragma unroll
    for (int c = 0; c < 4; ++c) {
        int j = t + c * 256;
        attn[(long)i * LSEQ + j] = (_Float16)(e[c] * inv);
    }
}

// ---------------------------------------------------------------------------
__global__ void cvt_f32_f16(const float* __restrict__ in, _Float16* __restrict__ out, long n) {
    long i = (long)blockIdx.x * blockDim.x + threadIdx.x;
    long stride = (long)gridDim.x * blockDim.x;
    for (; i < n; i += stride) out[i] = (_Float16)in[i];
}

// ---------------------------------------------------------------------------
// x[b][l][w] (f32) -> xT[b][w][l] (f16), 32x32 LDS tiles. block (32,8).
// ---------------------------------------------------------------------------
__global__ __launch_bounds__(256) void transpose_cvt(const float* __restrict__ x,
                                                     _Float16* __restrict__ xT) {
    __shared__ float tile[32][33];
    const long base = (long)blockIdx.z * LSEQ * WDIM;
    const int w0 = blockIdx.x * 32, l0 = blockIdx.y * 32;
    const int tx = threadIdx.x, ty = threadIdx.y;
#pragma unroll
    for (int k = 0; k < 4; ++k)
        tile[ty + k * 8][tx] = x[base + (long)(l0 + ty + k * 8) * WDIM + w0 + tx];
    __syncthreads();
#pragma unroll
    for (int k = 0; k < 4; ++k)
        xT[base + (long)(w0 + ty + k * 8) * LSEQ + l0 + tx] = (_Float16)tile[tx][ty + k * 8];
}

// ---------------------------------------------------------------------------
// WMMA GEMM: C[m][n] = sum_k A[m][k] * B[n][k]   (B row-major [N][K], f16).
// Block tile 128x256, BK=32, double-buffered LDS fed by async global->LDS.
// 8 waves as 2x4; each wave owns 64x64 = 4x4 WMMA tiles (128 accum VGPRs).
// EPI 0: f32 store | EPI 1: +bias, quick_gelu, f16 store | EPI 2: +bias+res, f32.
// ---------------------------------------------------------------------------
template <int EPI>
__global__ __launch_bounds__(256) void gemm_wmma(
    const _Float16* __restrict__ A, long sA,
    const _Float16* __restrict__ Bm, long sB,
    int lda, int ldb, int K,
    const float* __restrict__ bias,
    const float* __restrict__ res, long sRes, int ldres,
    void* __restrict__ Cv, long sC, int ldc) {
    constexpr int BM = 128, BN = 256, BK = 32, LDT = BK + 8;
    __shared__ __align__(16) _Float16 sAt[2][BM * LDT];  // 20 KB
    __shared__ __align__(16) _Float16 sBt[2][BN * LDT];  // 40 KB

    const int bz = blockIdx.z;
    A  += (long)bz * sA;
    Bm += (long)bz * sB;
    const int m0 = blockIdx.y * BM;
    const int n0 = blockIdx.x * BN;

    const int t    = threadIdx.x;
    const int wave = t >> 5;
    const int lane = t & 31;
    const int wm = wave >> 2;  // 2 waves along M (64 rows each)
    const int wn = wave & 3;   // 4 waves along N (64 cols each)
    const int lr = lane & 15;
    const int lg = lane >> 4;

    v8f acc[4][4] = {};

    // Issue async global->LDS copies for one BK-slice (tile at k-offset kt).
    auto issue_tile = [&](int kt, int buf) {
        // A: 128x32 halves = 512 x 16B chunks -> 2 per thread
#pragma unroll
        for (int i = 0; i < 2; ++i) {
            int c = t + i * 256;
            int r = c >> 2, kc = (c & 3) * 8;
            async_b128((unsigned int)(uintptr_t)&sAt[buf][r * LDT + kc],
                       &A[(long)(m0 + r) * lda + kt + kc]);
        }
        // B: 256x32 halves = 1024 x 16B chunks -> 4 per thread
#pragma unroll
        for (int i = 0; i < 4; ++i) {
            int c = t + i * 256;
            int r = c >> 2, kc = (c & 3) * 8;
            async_b128((unsigned int)(uintptr_t)&sBt[buf][r * LDT + kc],
                       &Bm[(long)(n0 + r) * ldb + kt + kc]);
        }
    };

    issue_tile(0, 0);
    wait_async0();
    __syncthreads();

    const int nk = K / BK;
    for (int kt = 0; kt < nk; ++kt) {
        if (kt + 1 < nk) issue_tile((kt + 1) * BK, (kt + 1) & 1);

        const int buf = kt & 1;
        // Fragment loads per CDNA5 16-bit layout:
        //   halves 0-7  -> K = lg*8 .. +7 ; halves 8-15 -> K = 16 + lg*8 .. +7
        v16h af[4], bf[4];
#pragma unroll
        for (int i = 0; i < 4; ++i) {
            const _Float16* pa = &sAt[buf][(wm * 64 + i * 16 + lr) * LDT + lg * 8];
            ((v8h*)&af[i])[0] = *(const v8h*)pa;
            ((v8h*)&af[i])[1] = *(const v8h*)(pa + 16);
        }
#pragma unroll
        for (int j = 0; j < 4; ++j) {
            const _Float16* pb = &sBt[buf][(wn * 64 + j * 16 + lr) * LDT + lg * 8];
            ((v8h*)&bf[j])[0] = *(const v8h*)pb;
            ((v8h*)&bf[j])[1] = *(const v8h*)(pb + 16);
        }
#pragma unroll
        for (int i = 0; i < 4; ++i)
#pragma unroll
            for (int j = 0; j < 4; ++j)
                acc[i][j] = __builtin_amdgcn_wmma_f32_16x16x32_f16(
                    false, af[i], false, bf[j], (short)0, acc[i][j], false, false);

        wait_async0();
        __syncthreads();
    }

    // Epilogue. C/D: VGPR r holds M = r + lg*8, N = lane&15.
#pragma unroll
    for (int i = 0; i < 4; ++i) {
#pragma unroll
        for (int j = 0; j < 4; ++j) {
#pragma unroll
            for (int r = 0; r < 8; ++r) {
                const int row = m0 + wm * 64 + i * 16 + r + lg * 8;
                const int col = n0 + wn * 64 + j * 16 + lr;
                float v = acc[i][j][r];
                if constexpr (EPI == 0) {
                    float* C = (float*)Cv + (long)bz * sC;
                    C[(long)row * ldc + col] = v;
                } else if constexpr (EPI == 1) {
                    v += bias[col];
                    float g = v / (1.0f + __expf(-1.702f * v));  // quick_gelu
                    _Float16* C = (_Float16*)Cv + (long)bz * sC;
                    C[(long)row * ldc + col] = (_Float16)g;
                } else {
                    v += bias[col] + res[(long)bz * sRes + (long)row * ldres + col];
                    float* C = (float*)Cv + (long)bz * sC;
                    C[(long)row * ldc + col] = v;
                }
            }
        }
    }
}

// ---------------------------------------------------------------------------
// LayerNorm over W=1024 (fp32), optional f16 side output for next GEMM.
// ---------------------------------------------------------------------------
template <bool W16>
__global__ __launch_bounds__(256) void ln_kernel(const float* __restrict__ src,
                                                 const float* __restrict__ gw,
                                                 const float* __restrict__ gb,
                                                 float* __restrict__ dst32,
                                                 _Float16* __restrict__ dst16) {
    const long row = blockIdx.x;
    const float* p = src + row * WDIM;
    const int t = threadIdx.x;
    __shared__ float s1[256], s2[256];
    float v[4];
    float a = 0.0f, q = 0.0f;
#pragma unroll
    for (int c = 0; c < 4; ++c) {
        v[c] = p[t + c * 256];
        a += v[c];
        q += v[c] * v[c];
    }
    s1[t] = a; s2[t] = q;
    __syncthreads();
    for (int stp = 128; stp > 0; stp >>= 1) {
        if (t < stp) { s1[t] += s1[t + stp]; s2[t] += s2[t + stp]; }
        __syncthreads();
    }
    const float mean = s1[0] * (1.0f / WDIM);
    const float var  = s2[0] * (1.0f / WDIM) - mean * mean;
    const float rs   = rsqrtf(var + 1e-5f);
#pragma unroll
    for (int c = 0; c < 4; ++c) {
        int col = t + c * 256;
        float y = (v[c] - mean) * rs * gw[col] + gb[col];
        dst32[row * WDIM + col] = y;
        if constexpr (W16) dst16[row * WDIM + col] = (_Float16)y;
    }
}

// ---------------------------------------------------------------------------
extern "C" void kernel_launch(void* const* d_in, const int* in_sizes, int n_in,
                              void* d_out, int out_size, void* d_ws, size_t ws_size,
                              hipStream_t stream) {
    const float* x      = (const float*)d_in[0];
    const float* ln1_w  = (const float*)d_in[1];
    const float* ln1_b  = (const float*)d_in[2];
    const float* ln2_w  = (const float*)d_in[3];
    const float* ln2_b  = (const float*)d_in[4];
    const float* fc_w   = (const float*)d_in[5];
    const float* fc_b   = (const float*)d_in[6];
    const float* proj_w = (const float*)d_in[7];
    const float* proj_b = (const float*)d_in[8];

    const int Bn = in_sizes[0] / (LSEQ * WDIM);  // 16

    char* p = (char*)d_ws;
    _Float16* attn16 = (_Float16*)p; p += (size_t)LSEQ * LSEQ * 2;
    _Float16* xT     = (_Float16*)p; p += (size_t)Bn * LSEQ * WDIM * 2;
    _Float16* w1h    = (_Float16*)p; p += (size_t)FDIM * WDIM * 2;
    _Float16* w2h    = (_Float16*)p; p += (size_t)WDIM * FDIM * 2;
    float*    xtm32  = (float*)p;    p += (size_t)Bn * LSEQ * WDIM * 4;
    _Float16* xtm16  = (_Float16*)p; p += (size_t)Bn * LSEQ * WDIM * 2;
    _Float16* h16    = (_Float16*)p; p += (size_t)Bn * LSEQ * FDIM * 2;
    float*    mixed  = (float*)p;    p += (size_t)Bn * LSEQ * WDIM * 4;  // reused as y

    attn_kernel<<<LSEQ, 256, 0, stream>>>(attn16);
    cvt_f32_f16<<<2048, 256, 0, stream>>>(fc_w, w1h, (long)FDIM * WDIM);
    cvt_f32_f16<<<2048, 256, 0, stream>>>(proj_w, w2h, (long)WDIM * FDIM);
    transpose_cvt<<<dim3(WDIM / 32, LSEQ / 32, Bn), dim3(32, 8), 0, stream>>>(x, xT);

    // mixed[b] = attn @ x[b]
    gemm_wmma<0><<<dim3(WDIM / 256, LSEQ / 128, Bn), 256, 0, stream>>>(
        attn16, 0, xT, (long)WDIM * LSEQ, LSEQ, LSEQ, LSEQ,
        nullptr, nullptr, 0, 0, mixed, (long)LSEQ * WDIM, WDIM);

    // x_tm = LN1(mixed)
    ln_kernel<true><<<Bn * LSEQ, 256, 0, stream>>>(mixed, ln1_w, ln1_b, xtm32, xtm16);

    // h = quick_gelu(x_tm @ fc_w^T + fc_b)
    gemm_wmma<1><<<dim3(FDIM / 256, LSEQ / 128, Bn), 256, 0, stream>>>(
        xtm16, (long)LSEQ * WDIM, w1h, 0, WDIM, WDIM, WDIM,
        fc_b, nullptr, 0, 0, h16, (long)LSEQ * FDIM, FDIM);

    // y = h @ proj_w^T + proj_b + x_tm
    gemm_wmma<2><<<dim3(WDIM / 256, LSEQ / 128, Bn), 256, 0, stream>>>(
        h16, (long)LSEQ * FDIM, w2h, 0, FDIM, FDIM, FDIM,
        proj_b, xtm32, (long)LSEQ * WDIM, WDIM, mixed, (long)LSEQ * WDIM, WDIM);

    // out = LN2(y)
    ln_kernel<false><<<Bn * LSEQ, 256, 0, stream>>>(mixed, ln2_w, ln2_b, (float*)d_out, nullptr);

    (void)n_in; (void)out_size; (void)ws_size;
}
